// DendSeqNet_45930380263772
// MI455X (gfx1250) — matrix operand order repaired and checked by reading, not compiled, not run on hardware
//
#include <hip/hip_runtime.h>
#include <stdint.h>
#include <stddef.h>

// ---------------------------------------------------------------------------
// DendSeqNet forward on MI455X (gfx1250, wave32, WMMA)
//
// Dominant cost: per-timestep GEMM [128x2048]x[2048x2048] per hidden channel.
// Strategy: bf16 hi/lo split-GEMM on v_wmma_f32_16x16x32_bf16 (3 WMMAs per
// K=32 chunk ~ f32 accuracy at bf16-pipe throughput), fragments pre-packed
// into the ISA VGPR layout so the inner loop is b128 loads + WMMA only.
// Each wave owns TWO M-tiles sharing one B fragment pair: 6 WMMAs per
// 6 fragment loads per k-step (16 FLOP/byte from L2).
// ---------------------------------------------------------------------------

typedef __attribute__((ext_vector_type(16))) __bf16    v16bf;
typedef __attribute__((ext_vector_type(8)))  float     v8f;
typedef __attribute__((ext_vector_type(8)))  unsigned  v8u;

#define T_STEPS 64
#define BATCH   128
#define FS2     4096
#define H1      2048
#define HC      2
#define SPL1    2048
#define OC      4
#define SPL2    512
#define OUTN    10
#define KSTEPS  64      // SPL1 / 32
#define MT      8       // BATCH / 16  (M tiles)
#define NTILES  128     // H1 / 16     (N tiles)

#define APACK_DW ((size_t)HC*MT*KSTEPS*32*8)      //   262,144 dwords (1 MB)
#define BPACK_DW ((size_t)HC*NTILES*KSTEPS*32*8)  // 4,194,304 dwords (16 MB)

// d_ws layout (units: dwords / floats)
static constexpr size_t OFF_APACK_HI = 0;
static constexpr size_t OFF_APACK_LO = OFF_APACK_HI + APACK_DW;
static constexpr size_t OFF_BPACK_HI = OFF_APACK_LO + APACK_DW;
static constexpr size_t OFF_BPACK_LO = OFF_BPACK_HI + BPACK_DW;
static constexpr size_t OFF_VDH = OFF_BPACK_LO + BPACK_DW;           // [HC,B,H1]
static constexpr size_t OFF_IDH = OFF_VDH + (size_t)HC*BATCH*H1;     // [HC,B,H1]
static constexpr size_t OFF_VSH = OFF_IDH + (size_t)HC*BATCH*H1;     // [B,H1]
static constexpr size_t OFF_ISH = OFF_VSH + (size_t)BATCH*H1;        // [B,H1]
static constexpr size_t OFF_ZS  = OFF_ISH + (size_t)BATCH*H1;        // [B,H1] spikes
static constexpr size_t OFF_VDO = OFF_ZS  + (size_t)BATCH*H1;        // [OC,B,OUT]
static constexpr size_t OFF_IDO = OFF_VDO + (size_t)OC*BATCH*OUTN;
static constexpr size_t OFF_VSO = OFF_IDO + (size_t)OC*BATCH*OUTN;   // [B,OUT]
static constexpr size_t OFF_ISO = OFF_VSO + (size_t)BATCH*OUTN;
static constexpr size_t OFF_END = OFF_ISO + (size_t)BATCH*OUTN;
static constexpr size_t STATE_DW = OFF_END - OFF_VDH;

// ---- bf16 helpers (RNE) ----------------------------------------------------
__device__ __forceinline__ unsigned f2bf(float f) {
  unsigned u = __float_as_uint(f);
  return (u + 0x7FFFu + ((u >> 16) & 1u)) >> 16;
}
__device__ __forceinline__ float bf2f(unsigned h) {
  return __uint_as_float(h << 16);
}
// split x0,x1 into packed bf16 hi-plane dword and lo-plane (residual) dword
__device__ __forceinline__ void split2(float x0, float x1,
                                       unsigned* dhi, unsigned* dlo) {
  unsigned h0 = f2bf(x0), h1 = f2bf(x1);
  float    r0 = x0 - bf2f(h0), r1 = x1 - bf2f(h1);
  *dhi = h0 | (h1 << 16);
  *dlo = f2bf(r0) | (f2bf(r1) << 16);
}

// ---- zero init -------------------------------------------------------------
__global__ void dsn_zero_kernel(float* __restrict__ p, unsigned n) {
  unsigned i = blockIdx.x * blockDim.x + threadIdx.x;
  unsigned stride = gridDim.x * blockDim.x;
  for (; i < n; i += stride) p[i] = 0.0f;
}

// ---- pack x_t into A-fragment order (hi/lo bf16 planes) --------------------
// A 16x32 bf16 layout: lanes 0-15 M=lane, kb=0; lanes 16-31 M=lane&15, kb=8.
// VGPR j<4 : K = kb + 2j,(+1);  j>=4 : K = 16 + kb + 2(j-4),(+1)
__global__ void dsn_pack_x_kernel(const float* __restrict__ x_t,
                                  unsigned* __restrict__ ahi,
                                  unsigned* __restrict__ alo) {
  unsigned idx = blockIdx.x * blockDim.x + threadIdx.x;  // [c][mt][ks][lane][j]
  if (idx >= (unsigned)APACK_DW) return;
  int j    = idx & 7;
  int lane = (idx >> 3) & 31;
  int ks   = (idx >> 8) & 63;
  int mt   = (idx >> 14) & 7;
  int c    = (idx >> 17) & 1;
  int kb   = (lane >> 4) * 8;
  int kp   = (j < 4) ? (kb + 2 * j) : (16 + kb + 2 * (j - 4));
  int k    = ks * 32 + kp;
  int m    = mt * 16 + (lane & 15);
  const float* p = x_t + (size_t)m * FS2 + (size_t)c * SPL1 + k;  // k,k+1 contiguous
  split2(p[0], p[1], &ahi[idx], &alo[idx]);
}

// ---- pack w_hidden into B-fragment order (hi/lo bf16 planes) ---------------
// B 32x16 bf16 layout: lane N = lane&15; lanes 0-15 kb=0, lanes 16-31 kb=8;
// VGPR j<4 : K = kb + 2j,(+1);  j>=4 : K = 16 + kb + 2(j-4),(+1)
__global__ void dsn_pack_w_kernel(const float* __restrict__ w,
                                  unsigned* __restrict__ bhi,
                                  unsigned* __restrict__ blo) {
  unsigned idx = blockIdx.x * blockDim.x + threadIdx.x;  // [c][nt][ks][lane][j]
  if (idx >= (unsigned)BPACK_DW) return;
  int j    = idx & 7;
  int lane = (idx >> 3) & 31;
  int ks   = (idx >> 8) & 63;
  int nt   = (idx >> 14) & 127;
  int c    = (idx >> 21) & 1;
  int kb   = (lane >> 4) * 8;
  int k    = ks * 32 + ((j < 4) ? (kb + 2 * j) : (16 + kb + 2 * (j - 4)));
  int n    = nt * 16 + (lane & 15);
  const float* p = w + ((size_t)c * SPL1 + k) * H1 + n;  // k,k+1 stride H1 apart
  split2(p[0], p[H1], &bhi[idx], &blo[idx]);
}

// ---- hidden layer: split-GEMM + dendritic/somatic LIF (fused) --------------
// grid = NTILES workgroups (16 output cols each); 4 waves, each owning TWO
// 16-row M-tiles that share one B fragment pair -> 6 WMMAs : 6 loads / k-step.
__global__ __launch_bounds__(128, 1)
void dsn_hidden_kernel(const unsigned* __restrict__ ahi,
                       const unsigned* __restrict__ alo,
                       const unsigned* __restrict__ bhi,
                       const unsigned* __restrict__ blo,
                       float* __restrict__ ws) {
  const int lane = threadIdx.x & 31;
  const int wv   = threadIdx.x >> 5;   // 0..3
  const int mt0  = wv * 2;             // first M-tile of this wave
  const int mt1  = wv * 2 + 1;         // second M-tile
  const int nt   = blockIdx.x;         // N-tile per workgroup

  float* vdh = ws + OFF_VDH;
  float* idh = ws + OFF_IDH;
  float* vsh = ws + OFF_VSH;
  float* ish = ws + OFF_ISH;
  float* zsb = ws + OFF_ZS;

  const int ncol  = nt * 16 + (lane & 15);
  const int rhalf = (lane & 16) >> 1;  // +8 row offset for lanes 16-31 (C layout)

  float zsum0[8], zsum1[8];
#pragma unroll
  for (int r = 0; r < 8; ++r) { zsum0[r] = 0.0f; zsum1[r] = 0.0f; }

  for (int c = 0; c < HC; ++c) {
    v8f acc0 = {};
    v8f acc1 = {};
    const size_t ab0 = (size_t)(c * MT + mt0) * (KSTEPS * 256) + (size_t)lane * 8;
    const size_t ab1 = (size_t)(c * MT + mt1) * (KSTEPS * 256) + (size_t)lane * 8;
    const size_t bb  = (size_t)(c * NTILES + nt) * (KSTEPS * 256) + (size_t)lane * 8;
#pragma unroll 4
    for (int ks = 0; ks < KSTEPS; ++ks) {
      const size_t ao0 = ab0 + (size_t)ks * 256;
      const size_t ao1 = ab1 + (size_t)ks * 256;
      const size_t bo  = bb + (size_t)ks * 256;
      // unconditional speculative prefetch (branch-free; OOB is dropped)
      __builtin_prefetch(bhi + bo + 8 * 256, 0, 3);
      __builtin_prefetch(blo + bo + 8 * 256, 0, 3);
      __builtin_prefetch(ahi + ao0 + 8 * 256, 0, 3);
      __builtin_prefetch(alo + ao0 + 8 * 256, 0, 3);
      v16bf bH  = __builtin_bit_cast(v16bf, *(const v8u*)(bhi + bo));
      v16bf bL  = __builtin_bit_cast(v16bf, *(const v8u*)(blo + bo));
      v16bf a0H = __builtin_bit_cast(v16bf, *(const v8u*)(ahi + ao0));
      v16bf a0L = __builtin_bit_cast(v16bf, *(const v8u*)(alo + ao0));
      v16bf a1H = __builtin_bit_cast(v16bf, *(const v8u*)(ahi + ao1));
      v16bf a1L = __builtin_bit_cast(v16bf, *(const v8u*)(alo + ao1));
      // acc += AhiBhi + AhiBlo + AloBhi   (Ozaki 2-word split, ~f32 accuracy)
      acc0 = __builtin_amdgcn_wmma_f32_16x16x32_bf16(false, a0H, false, bH, (short)0, acc0, false, false);
      acc1 = __builtin_amdgcn_wmma_f32_16x16x32_bf16(false, a1H, false, bH, (short)0, acc1, false, false);
      acc0 = __builtin_amdgcn_wmma_f32_16x16x32_bf16(false, a0H, false, bL, (short)0, acc0, false, false);
      acc1 = __builtin_amdgcn_wmma_f32_16x16x32_bf16(false, a1H, false, bL, (short)0, acc1, false, false);
      acc0 = __builtin_amdgcn_wmma_f32_16x16x32_bf16(false, a0L, false, bH, (short)0, acc0, false, false);
      acc1 = __builtin_amdgcn_wmma_f32_16x16x32_bf16(false, a1L, false, bH, (short)0, acc1, false, false);
    }
    // dendritic LIF for channel c; C-layout: VGPR r -> row (r | r+8), lane&15 -> col
#pragma unroll
    for (int r = 0; r < 8; ++r) {
      {
        const int b = mt0 * 16 + r + rhalf;
        const size_t di = ((size_t)c * BATCH + b) * H1 + ncol;
        float v = vdh[di], ii = idh[di], cur = acc0[r];
        float vdec  = fmaf(0.1f, ii - v, v);          // v + DT*TAU_MEM_INV*(i - v)
        float spike = (vdec > 1.0f) ? 1.0f : 0.0f;    // heaviside(v_dec - V_TH)
        vdh[di] = (vdec > 1.0f) ? 0.0f : vdec;        // reset to V_RESET=0
        idh[di] = fmaf(0.8f, ii, cur);                // i*(1-DT*TAU_SYN_INV) + cur
        zsum0[r] += spike;
      }
      {
        const int b = mt1 * 16 + r + rhalf;
        const size_t di = ((size_t)c * BATCH + b) * H1 + ncol;
        float v = vdh[di], ii = idh[di], cur = acc1[r];
        float vdec  = fmaf(0.1f, ii - v, v);
        float spike = (vdec > 1.0f) ? 1.0f : 0.0f;
        vdh[di] = (vdec > 1.0f) ? 0.0f : vdec;
        idh[di] = fmaf(0.8f, ii, cur);
        zsum1[r] += spike;
      }
    }
  }
  // somatic LIF on channel-summed spikes; emit z_s
#pragma unroll
  for (int r = 0; r < 8; ++r) {
    {
      const int b = mt0 * 16 + r + rhalf;
      const size_t si = (size_t)b * H1 + ncol;
      float v = vsh[si], ii = ish[si];
      float vdec  = fmaf(0.1f, ii - v, v);
      float spike = (vdec > 1.0f) ? 1.0f : 0.0f;
      vsh[si] = (vdec > 1.0f) ? 0.0f : vdec;
      ish[si] = fmaf(0.8f, ii, zsum0[r]);
      zsb[si] = spike;
    }
    {
      const int b = mt1 * 16 + r + rhalf;
      const size_t si = (size_t)b * H1 + ncol;
      float v = vsh[si], ii = ish[si];
      float vdec  = fmaf(0.1f, ii - v, v);
      float spike = (vdec > 1.0f) ? 1.0f : 0.0f;
      vsh[si] = (vdec > 1.0f) ? 0.0f : vdec;
      ish[si] = fmaf(0.8f, ii, zsum1[r]);
      zsb[si] = spike;
    }
  }
}

// ---- output layer: 4x [128x512]x[512x10] + LIF + LI readout ----------------
__global__ void dsn_out_kernel(const float* __restrict__ w_out,
                               float* __restrict__ ws,
                               float* __restrict__ out_t) {
  int tid = blockIdx.x * blockDim.x + threadIdx.x;
  if (tid >= BATCH * OUTN) return;
  int b = tid / OUTN, o = tid % OUTN;

  const float* zs = ws + OFF_ZS;
  float* vdo = ws + OFF_VDO;
  float* ido = ws + OFF_IDO;
  float* vso = ws + OFF_VSO;
  float* iso = ws + OFF_ISO;

  float ssum = 0.0f;
  for (int c = 0; c < OC; ++c) {
    float cur = 0.0f;
    const float* zrow = zs + (size_t)b * H1 + (size_t)c * SPL2;
    const float* wc   = w_out + (size_t)c * SPL2 * OUTN + o;
#pragma unroll 8
    for (int i = 0; i < SPL2; ++i) cur = fmaf(zrow[i], wc[(size_t)i * OUTN], cur);
    const size_t di = ((size_t)c * BATCH + b) * OUTN + o;
    float v = vdo[di], ii = ido[di];
    float vdec  = fmaf(0.1f, ii - v, v);
    float spike = (vdec > 1.0f) ? 1.0f : 0.0f;
    vdo[di] = (vdec > 1.0f) ? 0.0f : vdec;
    ido[di] = fmaf(0.8f, ii, cur);
    ssum += spike;
  }
  const size_t si = (size_t)b * OUTN + o;
  float v = vso[si], ii = iso[si];
  float vnew = fmaf(0.1f, ii - v, v);               // leaky integrator
  vso[si] = vnew;
  iso[si] = fmaf(0.8f, ii, ssum);
  out_t[si] = vnew;                                 // voltages[t]
}

// ---------------------------------------------------------------------------
extern "C" void kernel_launch(void* const* d_in, const int* in_sizes, int n_in,
                              void* d_out, int out_size, void* d_ws, size_t ws_size,
                              hipStream_t stream) {
  (void)in_sizes; (void)n_in; (void)out_size; (void)ws_size;
  const float* x        = (const float*)d_in[0];  // [T,B,FS2]
  const float* w_hidden = (const float*)d_in[1];  // [HC,SPL1,H1]
  const float* w_out    = (const float*)d_in[2];  // [OC,SPL2,OUT]
  float*    out = (float*)d_out;                  // [T,B,OUT]
  float*    wsf = (float*)d_ws;
  unsigned* wsu = (unsigned*)d_ws;

  // zero all LIF/LI state + spike buffer
  dsn_zero_kernel<<<1024, 256, 0, stream>>>(wsf + OFF_VDH, (unsigned)STATE_DW);
  // pack weights into B-fragment hi/lo planes (once per launch; L2-resident after)
  dsn_pack_w_kernel<<<(unsigned)((BPACK_DW + 255) / 256), 256, 0, stream>>>(
      w_hidden, wsu + OFF_BPACK_HI, wsu + OFF_BPACK_LO);

  for (int t = 0; t < T_STEPS; ++t) {
    dsn_pack_x_kernel<<<(unsigned)((APACK_DW + 255) / 256), 256, 0, stream>>>(
        x + (size_t)t * BATCH * FS2, wsu + OFF_APACK_HI, wsu + OFF_APACK_LO);
    dsn_hidden_kernel<<<NTILES, 128, 0, stream>>>(
        wsu + OFF_APACK_HI, wsu + OFF_APACK_LO,
        wsu + OFF_BPACK_HI, wsu + OFF_BPACK_LO, wsf);
    dsn_out_kernel<<<(BATCH * OUTN + 255) / 256, 256, 0, stream>>>(
        w_out, wsf, out + (size_t)t * BATCH * OUTN);
  }
}
